// SPEECHCOMMANDSDEEPSPEECH_44624710205766
// MI455X (gfx1250) — matrix operand order repaired and checked by reading, not compiled, not run on hardware
//
#include <hip/hip_runtime.h>
#include <hip/hip_bf16.h>

// ---------------------------------------------------------------------------
// Types & helpers for CDNA5 WMMA (bf16 in, f32 accum), wave32
// ---------------------------------------------------------------------------
typedef __attribute__((ext_vector_type(16))) __bf16          v16bf;
typedef __attribute__((ext_vector_type(8)))  float           v8f;
typedef __attribute__((ext_vector_type(8)))  unsigned short  v8u;
typedef __attribute__((ext_vector_type(16))) unsigned short  v16u;

__device__ __forceinline__ unsigned short f2bf(float f) {
    unsigned int u = __builtin_bit_cast(unsigned int, f);
    unsigned int r = (u + 0x7FFFu + ((u >> 16) & 1u)) >> 16;
    return (unsigned short)r;
}
__device__ __forceinline__ float sigmoidf_(float x) {
    return 1.0f / (1.0f + __expf(-x));
}

// A fragment: 16(M) x 32(K) from a row-major LDS tile (ld must be mult of 8).
// Per lane this is two contiguous 8-half (16B) runs -> ds_load_b128 x2.
// lanes 0-15: M=lane, K {0..7,16..23}; lanes 16-31: M=lane-16, K {8..15,24..31}
__device__ __forceinline__ v16bf load_a_frag(const unsigned short* base, int ld) {
    int lane = threadIdx.x & 31;
    int r = lane & 15;
    int khi = (lane >> 4) << 3;
    const unsigned short* p = base + r * ld + khi;
    v8u lo = *(const v8u*)p;
    v8u hi = *(const v8u*)(p + 16);
    v16u u = __builtin_shufflevector(lo, hi, 0, 1, 2, 3, 4, 5, 6, 7,
                                     8, 9, 10, 11, 12, 13, 14, 15);
    return __builtin_bit_cast(v16bf, u);
}

// B fragment: 32(K) x 16(N) from a COLUMN-MAJOR (NxK) matrix.
// Per lane two contiguous 16B runs -> global_load_b128 x2.
// lanes 0-15: N=lane, K {0..7,16..23}; lanes 16-31: N=lane-16, K {8..15,24..31}
__device__ __forceinline__ v16bf load_b_frag_cm(const unsigned short* B, int ldk,
                                                int n0, int k0) {
    int lane = threadIdx.x & 31;
    const unsigned short* p =
        B + (size_t)(n0 + (lane & 15)) * ldk + k0 + ((lane >> 4) << 3);
    v8u lo = *(const v8u*)p;
    v8u hi = *(const v8u*)(p + 16);
    v16u u = __builtin_shufflevector(lo, hi, 0, 1, 2, 3, 4, 5, 6, 7,
                                     8, 9, 10, 11, 12, 13, 14, 15);
    return __builtin_bit_cast(v16bf, u);
}

__device__ __forceinline__ v8f wmma_bf16(v16bf a, v16bf b, v8f c) {
    return __builtin_amdgcn_wmma_f32_16x16x32_bf16(false, a, false, b, (short)0, c,
                                                   false, false);
}

// ---------------------------------------------------------------------------
// Constant builders (on-device; graph-capture safe). ALL matrices column-major.
// ---------------------------------------------------------------------------

// BD_T: 448(N) x 416(K) bf16, column-major DFT matrix with Hann folded in.
//  n 0..200   : cos(2*pi*k*n/400)*hann(k)
//  n 224..424 : sin(2*pi*k*(n-224)/400)*hann(k); k>=400 and other n are zero.
__global__ void k_build_bd(unsigned short* BD) {
    int idx = blockIdx.x * 256 + threadIdx.x;
    if (idx >= 448 * 416) return;
    int n = idx / 416, k = idx % 416;
    float val = 0.0f;
    if (k < 400) {
        const float w = 6.283185307179586f / 400.0f;
        float hann = 0.5f - 0.5f * __cosf((float)k * w);
        int nn = -1; int is_sin = 0;
        if (n < 201) { nn = n; }
        else if (n >= 224 && n < 425) { nn = n - 224; is_sin = 1; }
        if (nn >= 0) {
            int ph = (k * nn) % 400;               // exact phase reduction
            float ang = (float)ph * w;
            val = (is_sin ? __sinf(ang) : __cosf(ang)) * hann;
        }
    }
    BD[idx] = f2bf(val);
}

__device__ __forceinline__ float mel_fpt(int i) {
    float melmax = 2595.0f * log10f(1.0f + 8000.0f / 700.0f);
    float m = melmax * (float)i / 129.0f;
    return 700.0f * (powf(10.0f, m / 2595.0f) - 1.0f);
}

// FB_T: 128(N) x 224(K) bf16 mel filterbank, column-major (k>=201 zero)
__global__ void k_build_fb(unsigned short* FB) {
    int idx = blockIdx.x * 256 + threadIdx.x;
    if (idx >= 128 * 224) return;
    int m = idx / 224, f = idx % 224;
    float val = 0.0f;
    if (f <= 200) {
        float freq = 40.0f * (float)f;
        float f0 = mel_fpt(m), f1 = mel_fpt(m + 1), f2 = mel_fpt(m + 2);
        float down = (freq - f0) / (f1 - f0);
        float up   = (f2 - freq) / (f2 - f1);
        val = fmaxf(0.0f, fminf(down, up));
    }
    FB[idx] = f2bf(val);
}

// WC_T: 128(N=out) x 384(K=3*128) bf16 from conv_w (out,in,k): K idx = k3*128+i
__global__ void k_build_wc(const float* __restrict__ conv_w, unsigned short* WC) {
    int idx = blockIdx.x * 256 + threadIdx.x;
    if (idx >= 128 * 384) return;
    int o = idx / 384, kk = idx % 384;
    int k3 = kk / 128, i = kk % 128;
    WC[idx] = f2bf(conv_w[o * 384 + i * 3 + k3]);
}

// WU_T: 512(N=gates) x 256(K) bf16; K rows 0..127 = W, 128..255 = U
__global__ void k_build_wu(const float* __restrict__ W, const float* __restrict__ U,
                           unsigned short* WU) {
    int idx = blockIdx.x * 256 + threadIdx.x;
    if (idx >= 512 * 256) return;
    int c = idx / 256, r = idx % 256;
    float v = (r < 128) ? W[r * 512 + c] : U[(r - 128) * 512 + c];
    WU[idx] = f2bf(v);
}

// FC1_T: 128(N) x 128(K) bf16 from fc1_w (in,out) row-major
__global__ void k_build_fc1(const float* __restrict__ fc1_w, unsigned short* F) {
    int idx = blockIdx.x * 256 + threadIdx.x;
    if (idx >= 128 * 128) return;
    int n = idx / 128, k = idx % 128;
    F[idx] = f2bf(fc1_w[k * 128 + n]);
}

// PRJ_T: 48(N, rows>=35 zero) x 128(K) bf16 from proj_w (128x35)
__global__ void k_build_proj(const float* __restrict__ proj_w, unsigned short* P) {
    int idx = blockIdx.x * 256 + threadIdx.x;
    if (idx >= 48 * 128) return;
    int n = idx / 128, k = idx % 128;
    P[idx] = f2bf(n < 35 ? proj_w[k * 35 + n] : 0.0f);
}

// ---------------------------------------------------------------------------
// Kernel 1: fused framing(reflect-pad) + windowed DFT GEMM + |.|^2
//   x(B,16000) f32  ->  power(B*81, 224) bf16
// Wave (m16 = w&3, pg = w>>2) computes 7 paired (Re,Im) 16x16 tiles so the
// power epilogue is register-local.
// ---------------------------------------------------------------------------
__global__ __launch_bounds__(256)
void dft_power_kernel(const float* __restrict__ x, const unsigned short* __restrict__ BD,
                      unsigned short* __restrict__ power) {
    __shared__ unsigned short Als[64 * 32];
    const int tid = threadIdx.x;
    const int wave = tid >> 5, lane = tid & 31;
    const int m16 = wave & 3, pg = wave >> 2;
    const int rows0 = blockIdx.x * 64;

    v8f accRe[7], accIm[7];
#pragma unroll
    for (int p = 0; p < 7; ++p) { accRe[p] = (v8f)0.0f; accIm[p] = (v8f)0.0f; }

    for (int k0 = 0; k0 < 416; k0 += 32) {
#pragma unroll
        for (int i = 0; i < 8; ++i) {
            int flat = i * 256 + tid;
            int r = flat >> 5, kk = flat & 31;
            int m = rows0 + r;
            int b = m / 81;
            int f = m - b * 81;
            int j = k0 + kk;
            float val = 0.0f;
            if (j < 400) {
                int s = f * 200 + j - 200;          // reflect padding
                if (s < 0) s = -s;
                if (s >= 16000) s = 31998 - s;
                val = x[b * 16000 + s];
            }
            Als[flat] = f2bf(val);
        }
        __syncthreads();
        v16bf a = load_a_frag(&Als[m16 * 16 * 32], 32);
#pragma unroll
        for (int p = 0; p < 7; ++p) {
            int pair = pg * 7 + p;
            v16bf bRe = load_b_frag_cm(BD, 416, pair * 16, k0);
            accRe[p] = wmma_bf16(a, bRe, accRe[p]);
            v16bf bIm = load_b_frag_cm(BD, 416, 224 + pair * 16, k0);
            accIm[p] = wmma_bf16(a, bIm, accIm[p]);
        }
        __syncthreads();
    }
#pragma unroll
    for (int p = 0; p < 7; ++p) {
        int pair = pg * 7 + p;
        int col = pair * 16 + (lane & 15);
#pragma unroll
        for (int r = 0; r < 8; ++r) {
            int row = rows0 + m16 * 16 + ((lane >> 4) << 3) + r;
            float re = accRe[p][r], im = accIm[p][r];
            power[(long)row * 224 + col] = f2bf(re * re + im * im);
        }
    }
}

// ---------------------------------------------------------------------------
// Generic bf16 GEMM:  C(MxN bf16) = A(MxK bf16) @ B_cm(NxK bf16) [+ bias]
// Block tile: M=128, N=64; 8 waves, each wave 16(M)x64(N). B loaded straight
// from global (column-major, L2-hot constants) as contiguous 32B runs.
// mode 0: A row base = row*K
// mode 1: (conv-as-GEMM) A row base = ((row/79)*81 + row%79)*128, K=384 contig
// ---------------------------------------------------------------------------
__global__ __launch_bounds__(256)
void gemm_bf16_kernel(const unsigned short* __restrict__ A,
                      const unsigned short* __restrict__ Bcm,
                      const float* __restrict__ bias,
                      unsigned short* __restrict__ Cm,
                      int N, int K, int mode) {
    __shared__ unsigned short Asm[128 * 32];
    const int tid = threadIdx.x;
    const int wave = tid >> 5, lane = tid & 31;
    const int rows0 = blockIdx.x * 128;
    const int cols0 = blockIdx.y * 64;

    v8f acc[4];
#pragma unroll
    for (int j = 0; j < 4; ++j) acc[j] = (v8f)0.0f;

    for (int k0 = 0; k0 < K; k0 += 32) {
        // stage A tile (128x32) as 16B chunks: 512 chunks / 256 threads
#pragma unroll
        for (int i = 0; i < 2; ++i) {
            int c = i * 256 + tid;
            int r = c >> 2, kc = (c & 3) * 8;
            int row = rows0 + r;
            long base;
            if (mode == 1) {
                int b = row / 79, t = row - b * 79;
                base = ((long)(b * 81 + t)) * 128;
            } else {
                base = (long)row * K;
            }
            *(v8u*)&Asm[r * 32 + kc] = *(const v8u*)&A[base + k0 + kc];
        }
        __syncthreads();
        v16bf a = load_a_frag(&Asm[wave * 16 * 32], 32);
#pragma unroll
        for (int j = 0; j < 4; ++j) {
            v16bf bf = load_b_frag_cm(Bcm, K, cols0 + j * 16, k0);
            acc[j] = wmma_bf16(a, bf, acc[j]);
        }
        __syncthreads();
    }
#pragma unroll
    for (int j = 0; j < 4; ++j) {
        int col = cols0 + j * 16 + (lane & 15);
        float bv = bias ? bias[col] : 0.0f;
#pragma unroll
        for (int r = 0; r < 8; ++r) {
            int row = rows0 + wave * 16 + ((lane >> 4) << 3) + r;
            Cm[(long)row * N + col] = f2bf(acc[j][r] + bv);
        }
    }
}

// ---------------------------------------------------------------------------
// LSTM layer: 64 batch rows per workgroup; h,c live in LDS; time sequential.
// Per step: gates(64x512) = [x_t | h](64x256) @ WU(256x512) + b, via WMMA.
// Wave (m16 = w&3, uh = w>>2): rows m16*16..+15, units uh*64..+63, ALL 4 gates
// -> i,f,g,o for one (row,unit) land in the same lane (no cross-wave traffic).
// Optionally stores h_t (bf16) and/or accumulates time-mean hbar (f32).
// ---------------------------------------------------------------------------
__global__ __launch_bounds__(256)
void lstm_kernel(const unsigned short* __restrict__ xin,
                 const unsigned short* __restrict__ WU,   // column-major 512x256
                 const float* __restrict__ bvec,
                 unsigned short* __restrict__ hout,       // nullable: (B,79,128) bf16
                 float* __restrict__ hbar) {              // nullable: (B,128) f32 mean
    __shared__ unsigned short Als[64 * 256];  // [x_t | h] tile, bf16, 32KB
    __shared__ float Cls[64 * 128];           // cell state, f32, 32KB
    const int tid = threadIdx.x;
    const int wave = tid >> 5, lane = tid & 31;
    const int m16 = wave & 3, uh = wave >> 2;
    const int blockRow = blockIdx.x * 64;

    // init h=0 (A-tile right half), c=0
    for (int i = 0; i < 64; ++i) Als[i * 256 + tid] = 0;
    for (int i = 0; i < 32; ++i) Cls[i * 256 + tid] = 0.0f;

    // per-lane bias for the 16 columns this lane touches
    float bb[4][4];
#pragma unroll
    for (int g = 0; g < 4; ++g)
#pragma unroll
        for (int j = 0; j < 4; ++j)
            bb[g][j] = bvec[g * 128 + uh * 64 + j * 16 + (lane & 15)];

    float hs[4][8];
#pragma unroll
    for (int j = 0; j < 4; ++j)
#pragma unroll
        for (int r = 0; r < 8; ++r) hs[j][r] = 0.0f;

    __syncthreads();

    for (int t = 0; t < 79; ++t) {
        // stage x_t into left half of A tile: 64x128 halves as 16B chunks
#pragma unroll
        for (int i = 0; i < 4; ++i) {
            int c = i * 256 + tid;              // 1024 chunks total
            int r = c >> 4, kc = (c & 15) * 8;
            *(v8u*)&Als[r * 256 + kc] =
                *(const v8u*)&xin[((long)(blockRow + r) * 79 + t) * 128 + kc];
        }
        __syncthreads();

        v8f acc[4][4];
#pragma unroll
        for (int g = 0; g < 4; ++g)
#pragma unroll
            for (int j = 0; j < 4; ++j) acc[g][j] = (v8f)0.0f;

        for (int k0 = 0; k0 < 256; k0 += 32) {
            v16bf a = load_a_frag(&Als[m16 * 16 * 256 + k0], 256);
#pragma unroll
            for (int g = 0; g < 4; ++g)
#pragma unroll
                for (int j = 0; j < 4; ++j) {
                    v16bf bf =
                        load_b_frag_cm(WU, 256, g * 128 + uh * 64 + j * 16, k0);
                    acc[g][j] = wmma_bf16(a, bf, acc[g][j]);
                }
        }
        __syncthreads();  // all A-tile reads done before h is overwritten

        // gate nonlinearity + state update
#pragma unroll
        for (int j = 0; j < 4; ++j) {
            int unit = uh * 64 + j * 16 + (lane & 15);
#pragma unroll
            for (int r = 0; r < 8; ++r) {
                int row = m16 * 16 + ((lane >> 4) << 3) + r;
                float iv = sigmoidf_(acc[0][j][r] + bb[0][j]);
                float fv = sigmoidf_(acc[1][j][r] + bb[1][j]);
                float gv = tanhf(acc[2][j][r] + bb[2][j]);
                float ov = sigmoidf_(acc[3][j][r] + bb[3][j]);
                float c = fv * Cls[row * 128 + unit] + iv * gv;
                Cls[row * 128 + unit] = c;
                float h = ov * tanhf(c);
                hs[j][r] += h;
                Als[row * 256 + 128 + unit] = f2bf(h);
                if (hout)
                    hout[((long)(blockRow + row) * 79 + t) * 128 + unit] = f2bf(h);
            }
        }
        __syncthreads();  // h visible before next step's WMMA
    }

    if (hbar) {
#pragma unroll
        for (int j = 0; j < 4; ++j) {
            int unit = uh * 64 + j * 16 + (lane & 15);
#pragma unroll
            for (int r = 0; r < 8; ++r) {
                int row = m16 * 16 + ((lane >> 4) << 3) + r;
                hbar[(long)(blockRow + row) * 128 + unit] = hs[j][r] * (1.0f / 79.0f);
            }
        }
    }
}

// ---------------------------------------------------------------------------
// Head: out = (hbar @ fc1 + fc1_b) @ proj + proj_b
// 128 rows per workgroup; two chained WMMA GEMMs through LDS.
// ---------------------------------------------------------------------------
__global__ __launch_bounds__(256)
void head_kernel(const float* __restrict__ hbar,
                 const unsigned short* __restrict__ FC1,   // cm 128x128
                 const float* __restrict__ fc1_b,
                 const unsigned short* __restrict__ PRJ,   // cm 48x128
                 const float* __restrict__ proj_b,
                 float* __restrict__ out) {
    __shared__ unsigned short Asm[128 * 128];  // 32KB
    __shared__ unsigned short Zsm[128 * 128];  // 32KB
    const int tid = threadIdx.x;
    const int wave = tid >> 5, lane = tid & 31;
    const int rows0 = blockIdx.x * 128;

#pragma unroll
    for (int i = 0; i < 64; ++i) {
        int flat = i * 256 + tid;
        int r = flat >> 7, k = flat & 127;
        Asm[flat] = f2bf(hbar[(long)(rows0 + r) * 128 + k]);
    }
    __syncthreads();

    // GEMM 1: Z = A @ FC1 + fc1_b   (wave computes 16 x 128)
    v8f acc1[8];
#pragma unroll
    for (int j = 0; j < 8; ++j) acc1[j] = (v8f)0.0f;
    for (int k0 = 0; k0 < 128; k0 += 32) {
        v16bf a = load_a_frag(&Asm[wave * 16 * 128 + k0], 128);
#pragma unroll
        for (int j = 0; j < 8; ++j) {
            v16bf bf = load_b_frag_cm(FC1, 128, j * 16, k0);
            acc1[j] = wmma_bf16(a, bf, acc1[j]);
        }
    }
#pragma unroll
    for (int j = 0; j < 8; ++j) {
        int col = j * 16 + (lane & 15);
        float bv = fc1_b[col];
#pragma unroll
        for (int r = 0; r < 8; ++r) {
            int row = wave * 16 + ((lane >> 4) << 3) + r;
            Zsm[row * 128 + col] = f2bf(acc1[j][r] + bv);
        }
    }
    __syncthreads();

    // GEMM 2: out = Z @ PROJ + proj_b   (N padded to 48, cols>=35 dropped)
    v8f acc2[3];
#pragma unroll
    for (int j = 0; j < 3; ++j) acc2[j] = (v8f)0.0f;
    for (int k0 = 0; k0 < 128; k0 += 32) {
        v16bf a = load_a_frag(&Zsm[wave * 16 * 128 + k0], 128);
#pragma unroll
        for (int j = 0; j < 3; ++j) {
            v16bf bf = load_b_frag_cm(PRJ, 128, j * 16, k0);
            acc2[j] = wmma_bf16(a, bf, acc2[j]);
        }
    }
#pragma unroll
    for (int j = 0; j < 3; ++j) {
        int col = j * 16 + (lane & 15);
        if (col < 35) {
            float bv = proj_b[col];
#pragma unroll
            for (int r = 0; r < 8; ++r) {
                int row = rows0 + wave * 16 + ((lane >> 4) << 3) + r;
                out[(long)row * 35 + col] = acc2[j][r] + bv;
            }
        }
    }
}

// ---------------------------------------------------------------------------
// Host-side launch orchestration
// ---------------------------------------------------------------------------
extern "C" void kernel_launch(void* const* d_in, const int* in_sizes, int n_in,
                              void* d_out, int out_size, void* d_ws, size_t ws_size,
                              hipStream_t stream) {
    (void)n_in; (void)out_size; (void)ws_size;
    const float* x      = (const float*)d_in[0];
    const float* conv_w = (const float*)d_in[1];
    const float* conv_b = (const float*)d_in[2];
    const float* W1     = (const float*)d_in[3];
    const float* U1     = (const float*)d_in[4];
    const float* b1     = (const float*)d_in[5];
    const float* W2     = (const float*)d_in[6];
    const float* U2     = (const float*)d_in[7];
    const float* b2     = (const float*)d_in[8];
    const float* fc1_w  = (const float*)d_in[9];
    const float* fc1_b  = (const float*)d_in[10];
    const float* proj_w = (const float*)d_in[11];
    const float* proj_b = (const float*)d_in[12];
    float* out = (float*)d_out;

    const int Bn = in_sizes[0] / 16000;  // 1024
    const int M1 = Bn * 81;              // 82944 frame rows
    const int Mc = Bn * 79;              // 80896 conv/time rows

    char* ws = (char*)d_ws;
    size_t off = 0;
    auto take = [&](size_t bytes) -> char* {
        char* p = ws + off;
        off = (off + bytes + 255) & ~(size_t)255;
        return p;
    };
    unsigned short* BD   = (unsigned short*)take((size_t)448 * 416 * 2);
    unsigned short* FB   = (unsigned short*)take((size_t)128 * 224 * 2);
    unsigned short* WC   = (unsigned short*)take((size_t)128 * 384 * 2);
    unsigned short* WU1  = (unsigned short*)take((size_t)512 * 256 * 2);
    unsigned short* WU2  = (unsigned short*)take((size_t)512 * 256 * 2);
    unsigned short* FC1  = (unsigned short*)take((size_t)128 * 128 * 2);
    unsigned short* PRJ  = (unsigned short*)take((size_t)48 * 128 * 2);
    unsigned short* PW   = (unsigned short*)take((size_t)M1 * 224 * 2);
    unsigned short* MEL  = (unsigned short*)take((size_t)M1 * 128 * 2);
    unsigned short* CVH  = (unsigned short*)take((size_t)Mc * 128 * 2);
    unsigned short* H1   = (unsigned short*)take((size_t)Mc * 128 * 2);
    float*          HBAR = (float*)take((size_t)Bn * 128 * 4);

    // constants (pure device-side, deterministic), all column-major
    k_build_bd  <<<(448 * 416 + 255) / 256, 256, 0, stream>>>(BD);
    k_build_fb  <<<(128 * 224 + 255) / 256, 256, 0, stream>>>(FB);
    k_build_wc  <<<(128 * 384 + 255) / 256, 256, 0, stream>>>(conv_w, WC);
    k_build_wu  <<<(512 * 256 + 255) / 256, 256, 0, stream>>>(W1, U1, WU1);
    k_build_wu  <<<(512 * 256 + 255) / 256, 256, 0, stream>>>(W2, U2, WU2);
    k_build_fc1 <<<(128 * 128 + 255) / 256, 256, 0, stream>>>(fc1_w, FC1);
    k_build_proj<<<(48 * 128 + 255) / 256, 256, 0, stream>>>(proj_w, PRJ);

    // mel spectrogram: framing + windowed DFT + power (WMMA)
    dft_power_kernel<<<M1 / 64, 256, 0, stream>>>(x, BD, PW);

    // power @ mel filterbank -> mel (B,81,128)
    dim3 g2(M1 / 128, 2);
    gemm_bf16_kernel<<<g2, 256, 0, stream>>>(PW, FB, nullptr, MEL, 128, 224, 0);

    // conv1d(k=3) as GEMM with row remap, + bias -> (B,79,128)
    dim3 g3(Mc / 128, 2);
    gemm_bf16_kernel<<<g3, 256, 0, stream>>>(MEL, WC, conv_b, CVH, 128, 384, 1);

    // LSTM layer 1 (stores per-step h), LSTM layer 2 (accumulates time-mean)
    lstm_kernel<<<Bn / 64, 256, 0, stream>>>(CVH, WU1, b1, H1, nullptr);
    lstm_kernel<<<Bn / 64, 256, 0, stream>>>(H1, WU2, b2, nullptr, HBAR);

    // (mean . fc1) folded: out = (hbar @ fc1 + b) @ proj + b
    head_kernel<<<Bn / 128, 256, 0, stream>>>(HBAR, FC1, fc1_b, PRJ, proj_b, out);
}
